// HANLayer_25606595019110
// MI455X (gfx1250) — compile-verified, matrix operand order checked by reading, not run on hardware
//
#include <hip/hip_runtime.h>
#include <hip/hip_bf16.h>
#include <math.h>

// ---------------------------------------------------------------------------
// HANLayer on MI455X (gfx1250): WMMA f16 GEMMs + atomic edge-softmax phases.
// ---------------------------------------------------------------------------

typedef __attribute__((ext_vector_type(16))) _Float16 v16h;
typedef __attribute__((ext_vector_type(8)))  float    v8f;

#define NN   50000      // nodes
#define EE   400000     // edges
#define DIN  256
#define HIDD 128
#define HDD  256        // H*D

// ---------------------------------------------------------------------------
// small utility kernels
// ---------------------------------------------------------------------------
__global__ void fill_kernel(float* __restrict__ p, float v, long n) {
    long i = (long)blockIdx.x * blockDim.x + threadIdx.x;
    long s = (long)gridDim.x * blockDim.x;
    for (; i < n; i += s) p[i] = v;
}

__global__ void cvt_f16_kernel(const float* __restrict__ in, _Float16* __restrict__ out, int n) {
    int i = blockIdx.x * blockDim.x + threadIdx.x;
    int s = gridDim.x * blockDim.x;
    for (; i < n; i += s) out[i] = (_Float16)in[i];
}

// ---------------------------------------------------------------------------
// WMMA GEMM: C[Nrows x MFULL] = A[Nrows x K] * B[K x MFULL]
// block = 256 threads (8 waves). Each block: 128 rows x 128 cols (blockIdx.y
// selects a 128-col slab). Wave w owns row strip [16w,16w+16). K step = 32.
// LDS tiles are stored pre-swizzled into the CDNA5 16-bit WMMA fragment
// layout so each lane's fragment = one contiguous 32B chunk.
//   A frag (16x32, f16): lane = m + 16*hi, elem e: k = (e&7)+(e>=8?16:0)+8*hi
//   inverse used at staging: hi=(k>>3)&1, e=(k&7)+((k>>4)<<3)
// ---------------------------------------------------------------------------
template<int K, int MFULL, bool A_F16, bool OUT_F16>
__global__ __launch_bounds__(256) void wmma_gemm_kernel(const void* __restrict__ Aptr,
                                                        const _Float16* __restrict__ B,
                                                        void* __restrict__ Cptr,
                                                        int Nrows) {
    __shared__ __align__(32) _Float16 Alds[8 * 512];
    __shared__ __align__(32) _Float16 Blds[8 * 512];

    const int tid  = threadIdx.x;
    const int wave = tid >> 5;
    const int lane = tid & 31;
    const int rowbase   = blockIdx.x * 128;
    const int colOffset = blockIdx.y * 128;

    v8f zero = {};
    v8f acc[8];
#pragma unroll
    for (int j = 0; j < 8; ++j) acc[j] = zero;

    for (int kb = 0; kb < K; kb += 32) {
        // ---- stage A tile: 128 rows x 32 k, convert to f16, swizzle ----
        for (int idx = tid; idx < 128 * 32; idx += 256) {
            int rl = idx >> 5;          // local row 0..127
            int k  = idx & 31;
            int rg = rowbase + rl; if (rg >= Nrows) rg = Nrows - 1;
            float a;
            if (A_F16) a = (float)((const _Float16*)Aptr)[(long)rg * K + kb + k];
            else       a = ((const float*)Aptr)[(long)rg * K + kb + k];
            int hi = (k >> 3) & 1;
            int e  = (k & 7) + ((k >> 4) << 3);
            int strip = rl >> 4, m = rl & 15;
            Alds[strip * 512 + (m + 16 * hi) * 16 + e] = (_Float16)a;
        }
        // ---- stage B tile: 32 k x 128 cols, swizzle ----
        for (int idx = tid; idx < 32 * 128; idx += 256) {
            int k  = idx >> 7;          // 0..31
            int nl = idx & 127;
            _Float16 b = B[(long)(kb + k) * MFULL + colOffset + nl];
            int hi = (k >> 3) & 1;
            int e  = (k & 7) + ((k >> 4) << 3);
            int ct = nl >> 4, nn = nl & 15;
            Blds[ct * 512 + (nn + 16 * hi) * 16 + e] = b;
        }
        __syncthreads();

        if (kb + 32 < K) {  // prefetch next B tile (global_prefetch_b8)
            __builtin_prefetch(&B[(long)(kb + 32) * MFULL + colOffset + (tid & 127)], 0, 1);
        }

        v16h afrag = *reinterpret_cast<const v16h*>(&Alds[wave * 512 + lane * 16]);
#pragma unroll
        for (int j = 0; j < 8; ++j) {
            v16h bfrag = *reinterpret_cast<const v16h*>(&Blds[j * 512 + lane * 16]);
            acc[j] = __builtin_amdgcn_wmma_f32_16x16x32_f16(
                false, afrag, false, bfrag, (short)0, acc[j], false, false);
        }
        __syncthreads();
    }

    // ---- store: C layout VGPR r -> row r + 8*(lane>=16), col = lane&15 ----
#pragma unroll
    for (int j = 0; j < 8; ++j) {
        int col = colOffset + j * 16 + (lane & 15);
#pragma unroll
        for (int r = 0; r < 8; ++r) {
            int row = rowbase + wave * 16 + r + ((lane >> 4) << 3);
            if (row < Nrows) {
                if (OUT_F16) ((_Float16*)Cptr)[(long)row * MFULL + col] = (_Float16)acc[j][r];
                else         ((float*)Cptr)[(long)row * MFULL + col]    = acc[j][r];
            }
        }
    }
}

// ---------------------------------------------------------------------------
// el/er = sum_d f[n,h,d] * attn_{l,r}[h,d]  — one wave per node
// ---------------------------------------------------------------------------
__global__ __launch_bounds__(256) void attn_kernel(const float* __restrict__ f,
                                                   const float* __restrict__ al,
                                                   const float* __restrict__ ar,
                                                   float* __restrict__ el,
                                                   float* __restrict__ er, int n) {
    int wave = threadIdx.x >> 5, lane = threadIdx.x & 31;
    int node = blockIdx.x * 8 + wave;
    if (node >= n) return;
#pragma unroll
    for (int h = 0; h < 8; ++h) {
        float v  = f[(long)node * HDD + h * 32 + lane];
        float sl = v * al[h * 32 + lane];
        float sr = v * ar[h * 32 + lane];
#pragma unroll
        for (int off = 16; off > 0; off >>= 1) {
            sl += __shfl_down(sl, off, 32);
            sr += __shfl_down(sr, off, 32);
        }
        if (lane == 0) { el[node * 8 + h] = sl; er[node * 8 + h] = sr; }
    }
}

// ---------------------------------------------------------------------------
// edge softmax phases
// ---------------------------------------------------------------------------
__device__ inline void atomicMaxF(float* addr, float v) {
    if (v >= 0.f) atomicMax((int*)addr, __float_as_int(v));
    else          atomicMin((unsigned int*)addr, (unsigned int)__float_as_int(v));
}

__global__ void edge_max_kernel(const int* __restrict__ src, const int* __restrict__ dst,
                                const float* __restrict__ el, const float* __restrict__ er,
                                float* __restrict__ ebuf, float* __restrict__ m, int E) {
    int e = blockIdx.x * blockDim.x + threadIdx.x;
    if (e >= E) return;
    int s = src[e], d = dst[e];
#pragma unroll
    for (int h = 0; h < 8; ++h) {
        float x = el[s * 8 + h] + er[d * 8 + h];
        x = x > 0.f ? x : 0.2f * x;                 // leaky_relu(0.2)
        ebuf[(long)e * 8 + h] = x;
        atomicMaxF(&m[d * 8 + h], x);
    }
}

__global__ void m_fix_kernel(float* __restrict__ m, long n) {
    long i = (long)blockIdx.x * blockDim.x + threadIdx.x;
    long s = (long)gridDim.x * blockDim.x;
    for (; i < n; i += s) if (m[i] == -INFINITY) m[i] = 0.f;
}

__global__ void edge_exp_kernel(const int* __restrict__ dst, float* __restrict__ ebuf,
                                const float* __restrict__ m, float* __restrict__ den, int E) {
    int e = blockIdx.x * blockDim.x + threadIdx.x;
    if (e >= E) return;
    int d = dst[e];
#pragma unroll
    for (int h = 0; h < 8; ++h) {
        float x = __expf(ebuf[(long)e * 8 + h] - m[d * 8 + h]);
        ebuf[(long)e * 8 + h] = x;
        atomicAdd(&den[d * 8 + h], x);
    }
}

// one wave per edge; lane d handles channel h*32+d for each head h
__global__ __launch_bounds__(256) void edge_aggr_kernel(const int* __restrict__ src,
                                                        const int* __restrict__ dst,
                                                        const float* __restrict__ ebuf,
                                                        const float* __restrict__ den,
                                                        const float* __restrict__ f,
                                                        float* __restrict__ z, int E) {
    int wave = threadIdx.x >> 5, lane = threadIdx.x & 31;
    int e = blockIdx.x * 8 + wave;
    if (e >= E) return;
    int s = src[e], d = dst[e];
#pragma unroll
    for (int i = 0; i < 8; ++i) {
        float alpha = ebuf[(long)e * 8 + i] / den[d * 8 + i];
        atomicAdd(&z[(long)d * HDD + i * 32 + lane],
                  alpha * f[(long)s * HDD + i * 32 + lane]);
    }
}

__global__ void elu_bias_kernel(float* __restrict__ z, const float* __restrict__ bias, long n) {
    long i = (long)blockIdx.x * blockDim.x + threadIdx.x;
    long s = (long)gridDim.x * blockDim.x;
    for (; i < n; i += s) {
        float x = z[i] + bias[i & 255];
        z[i] = x > 0.f ? x : expm1f(x);             // elu
    }
}

// ---------------------------------------------------------------------------
// semantic attention: accum += sum_n tanh(z[n]@W1 + b1) @ W2  — wave per node
// ---------------------------------------------------------------------------
__global__ __launch_bounds__(256) void semantic_kernel(const float* __restrict__ z,
                                                       const float* __restrict__ W1,
                                                       const float* __restrict__ b1,
                                                       const float* __restrict__ W2,
                                                       float* __restrict__ accum, int n) {
    __shared__ float zs[8][HDD];
    int wave = threadIdx.x >> 5, lane = threadIdx.x & 31;
    int node  = blockIdx.x * 8 + wave;
    int nodec = node < n ? node : n - 1;
#pragma unroll
    for (int i = 0; i < 8; ++i) zs[wave][i * 32 + lane] = z[(long)nodec * HDD + i * 32 + lane];
    __syncthreads();
    float s = 0.f;
#pragma unroll
    for (int jj = 0; jj < 4; ++jj) {
        int j = jj * 32 + lane;
        float t = b1[j];
        for (int c = 0; c < HDD; ++c) t = fmaf(zs[wave][c], W1[c * HIDD + j], t);
        s += tanhf(t) * W2[j];
    }
#pragma unroll
    for (int off = 16; off > 0; off >>= 1) s += __shfl_down(s, off, 32);
    if (lane == 0 && node < n) atomicAdd(accum, s);
}

__global__ void beta_kernel(const float* __restrict__ accum, float* __restrict__ beta, float invN) {
    if (threadIdx.x == 0 && blockIdx.x == 0) {
        float w1 = accum[0] * invN, w2 = accum[1] * invN;
        float mx = fmaxf(w1, w2);
        float e1 = __expf(w1 - mx), e2 = __expf(w2 - mx);
        float s  = e1 + e2;
        beta[0] = e1 / s; beta[1] = e2 / s;
    }
}

__global__ void combine_kernel(const float* __restrict__ z1, const float* __restrict__ z2,
                               const float* __restrict__ beta, float* __restrict__ out, long n) {
    float b0 = beta[0], b1 = beta[1];
    long i = (long)blockIdx.x * blockDim.x + threadIdx.x;
    long s = (long)gridDim.x * blockDim.x;
    for (; i < n; i += s) out[i] = b0 * z1[i] + b1 * z2[i];
}

// ---------------------------------------------------------------------------
// orchestration
// ---------------------------------------------------------------------------
extern "C" void kernel_launch(void* const* d_in, const int* in_sizes, int n_in,
                              void* d_out, int out_size, void* d_ws, size_t ws_size,
                              hipStream_t stream) {
    const float* feat    = (const float*)d_in[0];
    const int*   src1    = (const int*)d_in[1];
    const int*   dst1    = (const int*)d_in[2];
    const int*   src2    = (const int*)d_in[3];
    const int*   dst2    = (const int*)d_in[4];
    const float* W_proj  = (const float*)d_in[5];
    const float* fc1     = (const float*)d_in[6];
    const float* attn_l1 = (const float*)d_in[7];
    const float* attn_r1 = (const float*)d_in[8];
    const float* bias1   = (const float*)d_in[9];
    const float* fc2     = (const float*)d_in[10];
    const float* attn_l2 = (const float*)d_in[11];
    const float* attn_r2 = (const float*)d_in[12];
    const float* bias2   = (const float*)d_in[13];
    const float* W1s     = (const float*)d_in[14];
    const float* b1s     = (const float*)d_in[15];
    const float* W2s     = (const float*)d_in[16];
    float* out = (float*)d_out;

    // ---- workspace layout (256B aligned) ----
    char* ws = (char*)d_ws;
    size_t off = 0;
    auto alloc = [&](size_t bytes) -> void* {
        void* p = ws + off;
        off += (bytes + 255) & ~(size_t)255;
        return p;
    };
    _Float16* Wp16   = (_Float16*)alloc((size_t)DIN * HIDD * 2);
    _Float16* fc1_16 = (_Float16*)alloc((size_t)HIDD * HDD * 2);
    _Float16* fc2_16 = (_Float16*)alloc((size_t)HIDD * HDD * 2);
    _Float16* h16    = (_Float16*)alloc((size_t)NN * HIDD * 2);
    float* f    = (float*)alloc((size_t)NN * HDD * 4);
    float* z1   = (float*)alloc((size_t)NN * HDD * 4);
    float* z2   = (float*)alloc((size_t)NN * HDD * 4);
    float* el   = (float*)alloc((size_t)NN * 8 * 4);
    float* er   = (float*)alloc((size_t)NN * 8 * 4);
    float* mbuf = (float*)alloc((size_t)NN * 8 * 4);
    float* den  = (float*)alloc((size_t)NN * 8 * 4);
    float* ebuf = (float*)alloc((size_t)EE * 8 * 4);
    float* acc  = (float*)alloc(16 * 4);   // [0]=s1 [1]=s2 [2..3]=beta

    const int rowBlocks = (NN + 127) / 128;          // 391
    const int nodeWaveBlocks = (NN + 7) / 8;         // 6250
    const int edgeBlocks = (EE + 255) / 256;         // 1563
    const int edgeWaveBlocks = (EE + 7) / 8;         // 50000

    // ---- weight conversion ----
    cvt_f16_kernel<<<64, 256, 0, stream>>>(W_proj, Wp16, DIN * HIDD);
    cvt_f16_kernel<<<64, 256, 0, stream>>>(fc1, fc1_16, HIDD * HDD);
    cvt_f16_kernel<<<64, 256, 0, stream>>>(fc2, fc2_16, HIDD * HDD);

    // ---- shared projection h = feat @ W_proj (store f16 for next GEMM) ----
    wmma_gemm_kernel<DIN, HIDD, false, true>
        <<<dim3(rowBlocks, 1), 256, 0, stream>>>(feat, Wp16, h16, NN);

    const int*   srcs[2]  = {src1, src2};
    const int*   dsts[2]  = {dst1, dst2};
    const _Float16* fcs[2] = {fc1_16, fc2_16};
    const float* als[2]   = {attn_l1, attn_l2};
    const float* ars[2]   = {attn_r1, attn_r2};
    const float* biases[2] = {bias1, bias2};
    float* zs[2] = {z1, z2};

    for (int p = 0; p < 2; ++p) {
        // f = h @ fc   (N x 256, f32)
        wmma_gemm_kernel<HIDD, HDD, true, false>
            <<<dim3(rowBlocks, 2), 256, 0, stream>>>(h16, fcs[p], f, NN);
        // attention scores
        attn_kernel<<<nodeWaveBlocks, 256, 0, stream>>>(f, als[p], ars[p], el, er, NN);
        // init segment buffers
        fill_kernel<<<512, 256, 0, stream>>>(mbuf, -INFINITY, (long)NN * 8);
        fill_kernel<<<512, 256, 0, stream>>>(den, 0.f, (long)NN * 8);
        fill_kernel<<<2048, 256, 0, stream>>>(zs[p], 0.f, (long)NN * HDD);
        // edge softmax
        edge_max_kernel<<<edgeBlocks, 256, 0, stream>>>(srcs[p], dsts[p], el, er, ebuf, mbuf, EE);
        m_fix_kernel<<<512, 256, 0, stream>>>(mbuf, (long)NN * 8);
        edge_exp_kernel<<<edgeBlocks, 256, 0, stream>>>(dsts[p], ebuf, mbuf, den, EE);
        // weighted aggregation
        edge_aggr_kernel<<<edgeWaveBlocks, 256, 0, stream>>>(srcs[p], dsts[p], ebuf, den, f, zs[p], EE);
        // bias + elu
        elu_bias_kernel<<<2048, 256, 0, stream>>>(zs[p], biases[p], (long)NN * HDD);
    }

    // ---- semantic attention ----
    fill_kernel<<<1, 32, 0, stream>>>(acc, 0.f, 4);
    semantic_kernel<<<nodeWaveBlocks, 256, 0, stream>>>(z1, W1s, b1s, W2s, &acc[0], NN);
    semantic_kernel<<<nodeWaveBlocks, 256, 0, stream>>>(z2, W1s, b1s, W2s, &acc[1], NN);
    beta_kernel<<<1, 1, 0, stream>>>(acc, &acc[2], 1.0f / (float)NN);
    combine_kernel<<<2048, 256, 0, stream>>>(z1, z2, &acc[2], out, (long)NN * HDD);

    (void)in_sizes; (void)n_in; (void)out_size; (void)ws_size;
}
